// SimpleSNN_61701500174804
// MI455X (gfx1250) — compile-verified, matrix operand order checked by reading, not run on hardware
//
#include <hip/hip_runtime.h>

// ---------------------------------------------------------------------------
// SNN 4-layer LIF forward, CDNA5 (gfx1250), wave32 WMMA f16 path.
//
// Orientation: all tiles are TRANSPOSED [feature(M) x batch(N)] so that a
// layer's f32 16x16 C/D tile pair is bit-position-identical to the next
// layer's f16 B operand (32xK x 16xN) after per-element f32->f16 convert:
//   B lane l<16 : N=l,    e0..7 -> K=e,    e8..15 -> K=8+e
//   B lane l>=16: N=l-16, e0..7 -> K=8+e,  e8..15 -> K=16+e
//   C tile-a (feats 0..15):  VGPR r, lane<16 -> M=r,   lane>=16 -> M=8+r
//   C tile-b (feats 16..31): same with M+16
// => B.e = (e<8) ? tileA[e] : tileB[e-8]   (no cross-lane movement).
// Weights are pre-swizzled once into A-operand v16h registers.
// ---------------------------------------------------------------------------

typedef _Float16 v16h __attribute__((ext_vector_type(16)));
typedef float    v8f  __attribute__((ext_vector_type(8)));
typedef float    v4f  __attribute__((ext_vector_type(4)));

#define BETA   0.9f
#define THRESH 1.0f
#define T_LEN  200

// Build one 16x32 f16 A-operand tile from a row-major [Fout x Fin] f32 weight
// matrix. mBase selects output-feature rows [mBase, mBase+16).
__device__ inline v16h make_wtile(const float* __restrict__ W, int Fout, int Fin,
                                  int mBase, int lane) {
  const bool hi = lane >= 16;
  const int  m  = mBase + (lane & 15);
  v16h w{};
#pragma unroll
  for (int e = 0; e < 16; ++e) {
    // A-layout (16-bit, 16x32): lane<16: e<8->K=e, e>=8->K=8+e
    //                           lane>=16: e<8->K=8+e, e>=8->K=16+e
    int K = (e < 8) ? (e + (hi ? 8 : 0)) : (e + (hi ? 16 : 8));
    float v = (m < Fout && K < Fin) ? W[m * Fin + K] : 0.0f;
    w[e] = (_Float16)v;
  }
  return w;
}

// Bias in C-tile layout: element r holds b[mBase + r + (hi?8:0)].
__device__ inline v8f make_bias(const float* __restrict__ b, int Fout,
                                int mBase, bool hi) {
  v8f r{};
#pragma unroll
  for (int i = 0; i < 8; ++i) {
    int f = mBase + i + (hi ? 8 : 0);
    r[i] = (f < Fout) ? b[f] : 0.0f;
  }
  return r;
}

// One LIF layer with 32 output features (two 16x16 tiles).
// bin: input spikes as f16 B operand (K=32 in-features x 16 batch).
// Returns this layer's spikes, already in B-operand layout for the next layer.
__device__ inline v16h lif32(const v16h& bin, const v16h& wa, const v16h& wb,
                             const v8f& biasA, const v8f& biasB,
                             v8f& ma, v8f& mb) {
  v8f preA = BETA * ma + biasA;           // mem decay + bias
  v8f preB = BETA * mb + biasB;
  v8f za = __builtin_amdgcn_wmma_f32_16x16x32_f16(false, wa, false, bin,
                                                  (short)0, preA, false, false);
  v8f zb = __builtin_amdgcn_wmma_f32_16x16x32_f16(false, wb, false, bin,
                                                  (short)0, preB, false, false);
  v16h s;
#pragma unroll
  for (int r = 0; r < 8; ++r) {
    float sa = (za[r] >= THRESH) ? 1.0f : 0.0f;   // hard spike (forward)
    float sb = (zb[r] >= THRESH) ? 1.0f : 0.0f;
    ma[r] = za[r] - sa;                            // reset-by-subtraction
    mb[r] = zb[r] - sb;
    s[r]     = (_Float16)sa;                       // C-pair -> next B operand
    s[r + 8] = (_Float16)sb;
  }
  return s;
}

__global__ __launch_bounds__(256) void
snn_lif_wmma_kernel(const float* __restrict__ x_seq,   // [T, B, 20]
                    const float* __restrict__ W1, const float* __restrict__ b1,
                    const float* __restrict__ W2, const float* __restrict__ b2,
                    const float* __restrict__ W3, const float* __restrict__ b3,
                    const float* __restrict__ W4, const float* __restrict__ b4,
                    float* __restrict__ out,            // [B, 10]
                    int batch) {
  const int lane = threadIdx.x & 31;
  const int wv   = threadIdx.x >> 5;
  const bool hi  = lane >= 16;
  const int mrow = lane & 15;                       // batch col within tile
  const int batchBase = (blockIdx.x * 8 + wv) * 16; // 16 batch rows per wave

  // ---- static weight A-tiles + biases (resident in VGPRs) ----
  const v16h w1a = make_wtile(W1, 32, 20, 0,  lane);
  const v16h w1b = make_wtile(W1, 32, 20, 16, lane);
  const v16h w2a = make_wtile(W2, 32, 32, 0,  lane);
  const v16h w2b = make_wtile(W2, 32, 32, 16, lane);
  const v16h w3a = make_wtile(W3, 32, 32, 0,  lane);
  const v16h w3b = make_wtile(W3, 32, 32, 16, lane);
  const v16h w4t = make_wtile(W4, 10, 32, 0,  lane); // rows 10..15 zero-padded

  const v8f bi1a = make_bias(b1, 32, 0, hi),  bi1b = make_bias(b1, 32, 16, hi);
  const v8f bi2a = make_bias(b2, 32, 0, hi),  bi2b = make_bias(b2, 32, 16, hi);
  const v8f bi3a = make_bias(b3, 32, 0, hi),  bi3b = make_bias(b3, 32, 16, hi);
  const v8f bi4  = make_bias(b4, 10, 0, hi);

  // ---- recurrent state ----
  v8f m1a{}, m1b{}, m2a{}, m2b{}, m3a{}, m3b{}, m4{};
  v8f cnt{};

  const size_t tStride = (size_t)batch * 20;        // floats per timestep
  const float* xrow = x_seq + ((size_t)batchBase + mrow) * 20;
  const int h0 = hi ? 8 : 0;                        // which 8 input cols

  for (int t = 0; t < T_LEN; ++t) {
    // x tile as B operand: lane<16 -> cols 0..7 & 16..19(+pad0);
    //                      lane>=16 -> cols 8..15 & pad0
    v4f a0 = *(const v4f*)(xrow + h0);
    v4f a1 = *(const v4f*)(xrow + h0 + 4);
    v4f a2 = *(const v4f*)(xrow + 16);              // discarded for hi lanes
    v16h bx;
#pragma unroll
    for (int e = 0; e < 4; ++e) {
      bx[e]      = (_Float16)a0[e];
      bx[e + 4]  = (_Float16)a1[e];
      bx[e + 8]  = hi ? (_Float16)0.0f : (_Float16)a2[e]; // cols 16..19 pad
      bx[e + 12] = (_Float16)0.0f;                        // cols 20..23 (lo) / 28..31 (hi) pad
    }

    // prefetch next timestep's row (stream, no reuse)
    const float* pf = (t + 1 < T_LEN) ? (xrow + tStride) : xrow;
    __builtin_prefetch((const void*)pf, 0, 0);

    // ---- 4 LIF layers: 7 WMMAs per timestep per wave ----
    v16h s1 = lif32(bx, w1a, w1b, bi1a, bi1b, m1a, m1b);
    v16h s2 = lif32(s1, w2a, w2b, bi2a, bi2b, m2a, m2b);
    v16h s3 = lif32(s2, w3a, w3b, bi3a, bi3b, m3a, m3b);

    // layer 4: single tile (10 valid output features)
    v8f pre4 = BETA * m4 + bi4;
    v8f z4 = __builtin_amdgcn_wmma_f32_16x16x32_f16(false, w4t, false, s3,
                                                    (short)0, pre4, false, false);
#pragma unroll
    for (int r = 0; r < 8; ++r) {
      float s = (z4[r] >= THRESH) ? 1.0f : 0.0f;
      m4[r]  = z4[r] - s;
      cnt[r] += s;                                   // spike count accumulate
    }

    xrow += tStride;
  }

  // ---- store counts: C tile (feat=r+8*hi, batch=mrow) -> out[B,10] ----
  float* orow = out + (size_t)(batchBase + mrow) * 10;
#pragma unroll
  for (int r = 0; r < 8; ++r) {
    int feat = r + (hi ? 8 : 0);
    if (feat < 10) orow[feat] = cnt[r];
  }
}

extern "C" void kernel_launch(void* const* d_in, const int* in_sizes, int n_in,
                              void* d_out, int out_size, void* d_ws, size_t ws_size,
                              hipStream_t stream) {
  (void)n_in; (void)d_ws; (void)ws_size; (void)in_sizes;
  const float* x_seq = (const float*)d_in[0];
  const float* W1 = (const float*)d_in[1]; const float* b1 = (const float*)d_in[2];
  const float* W2 = (const float*)d_in[3]; const float* b2 = (const float*)d_in[4];
  const float* W3 = (const float*)d_in[5]; const float* b3 = (const float*)d_in[6];
  const float* W4 = (const float*)d_in[7]; const float* b4 = (const float*)d_in[8];
  float* out = (float*)d_out;

  const int batch = out_size / 10;          // 32768
  const int blocks = batch / 128;           // 8 waves/block * 16 batch/wave
  snn_lif_wmma_kernel<<<blocks, 256, 0, stream>>>(x_seq, W1, b1, W2, b2,
                                                  W3, b3, W4, b4, out, batch);
}
